// CrossTransformer_46737834115494
// MI455X (gfx1250) — matrix operand order repaired and checked by reading, not compiled, not run on hardware
//
#include <hip/hip_runtime.h>

#define LDIM 256
#define DDIM 128

typedef __attribute__((ext_vector_type(16))) __bf16      v16bf;
typedef __attribute__((ext_vector_type(8)))  float       v8f;
typedef __attribute__((ext_vector_type(4)))  unsigned int u32x4;

struct U32x8 { u32x4 a, b; };

__device__ __forceinline__ v16bf pack16(u32x4 x, u32x4 y) {
  U32x8 t; t.a = x; t.b = y;
  return __builtin_bit_cast(v16bf, t);
}

// fp32 -> bf16 via native hardware convert (backend emits v_cvt_*bf16*)
__device__ __forceinline__ unsigned short f2bfu(float f) {
  __bf16 b = (__bf16)f;
  return __builtin_bit_cast(unsigned short, b);
}

__device__ __forceinline__ v8f v8zero() {
  v8f z;
#pragma unroll
  for (int i = 0; i < 8; ++i) z[i] = 0.f;
  return z;
}

// CDNA5 async global->LDS byte copy (16B per op), tracked with ASYNCcnt.
__device__ __forceinline__ void async_copy_b128(unsigned lds_addr, const void* gptr) {
  unsigned long long ga = (unsigned long long)gptr;
  asm volatile("global_load_async_to_lds_b128 %0, %1, off"
               :: "v"(lds_addr), "v"(ga) : "memory");
}
__device__ __forceinline__ void wait_async0() {
  asm volatile("s_wait_asynccnt 0" ::: "memory");
}

// A-matrix (16x32 bf16) register layout load; src row-major [M][K], 16B-aligned rows.
__device__ __forceinline__ v16bf load_a16(const unsigned short* src, int ld, int row0, int k0, int lane) {
  const int m = lane & 15, hi = (lane >> 4) & 1;
  const unsigned short* p = src + (size_t)(row0 + m) * ld + (k0 + hi * 8);
  return pack16(*(const u32x4*)(p), *(const u32x4*)(p + 16));
}

// B-matrix (32x16 bf16) load from N-major storage: element(k,n) = src[n*ld + k].
__device__ __forceinline__ v16bf load_bt16(const unsigned short* src, int ld, int k0, int n0, int lane) {
  const int n = lane & 15, hi = (lane >> 4) & 1;
  const unsigned short* p = src + (size_t)(n0 + n) * ld + (k0 + hi * 8);
  return pack16(*(const u32x4*)(p), *(const u32x4*)(p + 16));
}

__device__ __forceinline__ v16bf load_a16_pred(const unsigned short* base, int k0, int hi, bool valid) {
  if (!valid) { u32x4 z = {0u, 0u, 0u, 0u}; return pack16(z, z); }
  const unsigned short* p = base + k0 + hi * 8;
  return pack16(*(const u32x4*)(p), *(const u32x4*)(p + 16));
}

__device__ __forceinline__ v8f wmma_bf16(v16bf a, v16bf b, v8f c) {
  return __builtin_amdgcn_wmma_f32_16x16x32_bf16(false, a, false, b, (short)0, c, false, false);
}

// reductions across the 16-lane C-layout row group (xor of bits 0..3 stays in group)
__device__ __forceinline__ float grp16_max(float v) {
#pragma unroll
  for (int m = 1; m < 16; m <<= 1) v = fmaxf(v, __shfl_xor(v, m, 32));
  return v;
}
__device__ __forceinline__ float grp16_sum(float v) {
#pragma unroll
  for (int m = 1; m < 16; m <<= 1) v += __shfl_xor(v, m, 32);
  return v;
}

// ---------------------------------------------------------------------------
// K0: weight conversion to bf16, B operands stored N-major for load_bt16
// ---------------------------------------------------------------------------
__global__ void convert_weights_kernel(const float* __restrict__ hw, const float* __restrict__ vw,
                                       const float* __restrict__ dw, const float* __restrict__ c1,
                                       const float* __restrict__ c2,
                                       unsigned short* __restrict__ hwb, unsigned short* __restrict__ vwb,
                                       unsigned short* __restrict__ dwb, unsigned short* __restrict__ c1b,
                                       unsigned short* __restrict__ c2b) {
  int idx = blockIdx.x * 256 + threadIdx.x;
  if (idx < 49152) {  // (128,384) -> [n=384][k=128]
    int n = idx / 128, k = idx % 128;
    hwb[idx] = f2bfu(hw[(size_t)k * 384 + n]);
    vwb[idx] = f2bfu(vw[(size_t)k * 384 + n]);
  }
  if (idx < 32768) {  // (256,128) -> [n=128][k=256]
    int n = idx / 256, k = idx % 256;
    dwb[idx] = f2bfu(dw[(size_t)k * 128 + n]);
  }
  if (idx < 147456) { // (co,ci,3,3) -> [t=9][co=128][ci=128]
    int co = idx / 1152, rem = idx % 1152, ci = rem / 9, t = rem % 9;
    size_t o = ((size_t)t * 128 + co) * 128 + ci;
    c1b[o] = f2bfu(c1[idx]);
    c2b[o] = f2bfu(c2[idx]);
  }
}

// ---------------------------------------------------------------------------
// K0b: v_keep[n=(b,i)] = (sum_r mask[b,r,i]) != L
// ---------------------------------------------------------------------------
__global__ void colsum_kernel(const int* __restrict__ mask, int* __restrict__ keepflag) {
  int n = blockIdx.x * 256 + threadIdx.x;
  if (n >= 512) return;
  int b = n >> 8, i = n & 255, s = 0;
  for (int r = 0; r < LDIM; ++r) s += mask[(b * LDIM + r) * LDIM + i];
  keepflag[n] = (s != LDIM) ? 1 : 0;
}

// ---------------------------------------------------------------------------
// K1: xp = clip(x,±1000) + pos_embed_w[j>=i]
// ---------------------------------------------------------------------------
__global__ void prep_kernel(const float* __restrict__ x, const float* __restrict__ pe,
                            float* __restrict__ xp) {
  size_t idx = (size_t)blockIdx.x * 256 + threadIdx.x;
  if (idx >= (size_t)16777216) return;
  int d = (int)(idx & 127);
  size_t pix = idx >> 7;
  int j = (int)(pix & 255), i = (int)((pix >> 8) & 255);
  float v = x[idx];
  v = fminf(fmaxf(v, -1000.f), 1000.f);
  xp[idx] = v + pe[((j >= i) ? 1 : 0) * DDIM + d];
}

// ---------------------------------------------------------------------------
// K2: fused per-sequence attention (dir=0 horizontal, dir=1 vertical).
// One workgroup (8 wave32) per sequence; QKV GEMM + RoPE + full softmax
// attention in one LDS-resident pass. Writes bf16 into concat buffer.
// ---------------------------------------------------------------------------
__global__ __launch_bounds__(256) void attn_kernel(const float* __restrict__ xp,
                                                   const int* __restrict__ mask,
                                                   const unsigned short* __restrict__ Wbf,
                                                   const float* __restrict__ bias,
                                                   unsigned short* __restrict__ concat_out, int dir) {
  __shared__ unsigned short Xs[LDIM * DDIM];  // 64KB, recycled as P scratch
  __shared__ unsigned short Qs[LDIM * DDIM];  // 64KB [row][d]
  __shared__ unsigned short KsT[LDIM * DDIM]; // 64KB [key][d]
  __shared__ unsigned short VsT[DDIM * LDIM]; // 64KB [d][key]
  __shared__ float Mrow[LDIM];

  const int n = blockIdx.x;   // sequence in [0,512)
  const int b = n >> 8;
  const int rc = n & 255;     // row i (horiz) or column c (vert)
  const int tid = threadIdx.x, lane = tid & 31, wave = tid >> 5;
  const int nlo = lane & 15, hi = (lane >> 4) & 1;

  for (int idx = tid; idx < LDIM * DDIM; idx += 256) {
    int row = idx >> 7, d = idx & 127;
    size_t g = (dir == 0) ? ((size_t)n * LDIM + row) * DDIM + d
                          : (((size_t)(b * LDIM + row)) * LDIM + rc) * DDIM + d;
    Xs[idx] = f2bfu(xp[g]);
  }
  if (tid < LDIM)
    Mrow[tid] = (dir == 0) ? (float)mask[(b * LDIM + rc) * LDIM + tid]
                           : (float)mask[(b * LDIM + tid) * LDIM + rc];
  __syncthreads();

  // ---- QKV = clip(Xs @ W + bias, ±1e4); RoPE on Q,K.
  // Each wave owns row blocks {2w, 2w+1}; one B-tile load feeds both (2x reuse).
  {
    v16bf aX[2][4];
#pragma unroll
    for (int h = 0; h < 2; ++h)
#pragma unroll
      for (int kk = 0; kk < 4; ++kk)
        aX[h][kk] = load_a16(Xs, DDIM, (wave * 2 + h) * 16, kk * 32, lane);

    for (int nt = 0; nt < 24; ++nt) {
      float bv = bias[nt * 16 + nlo];
      v8f acc[2];
#pragma unroll
      for (int h = 0; h < 2; ++h)
#pragma unroll
        for (int r = 0; r < 8; ++r) acc[h][r] = bv;
#pragma unroll
      for (int kk = 0; kk < 4; ++kk) {
        v16bf bw = load_bt16(Wbf, DDIM, kk * 32, nt * 16, lane);
        acc[0] = wmma_bf16(aX[0][kk], bw, acc[0]);
        acc[1] = wmma_bf16(aX[1][kk], bw, acc[1]);
      }
#pragma unroll
      for (int h = 0; h < 2; ++h) {
        const int mb = wave * 2 + h;
#pragma unroll
        for (int r = 0; r < 8; ++r) acc[h][r] = fminf(fmaxf(acc[h][r], -1e4f), 1e4f);
        if (nt < 16) {  // Q (nt<8) or K (8..15): apply RoPE
          const int f = (nt & 7) * 16 + nlo;
          const float invf = __expf(-(float)(f & ~1) * (9.210340371976184f / 128.f));
#pragma unroll
          for (int r = 0; r < 8; ++r) {
            float pos = (float)(mb * 16 + r + 8 * hi);
            float sa, ca;
            __sincosf(pos * invf, &sa, &ca);
            float other = __shfl_xor(acc[h][r], 1, 32);  // partner feature f^1, same row
            acc[h][r] = (f & 1) ? (acc[h][r] * ca + other * sa) : (acc[h][r] * ca - other * sa);
          }
          unsigned short* dst = (nt < 8) ? Qs : KsT;
#pragma unroll
          for (int r = 0; r < 8; ++r) {
            int row = mb * 16 + r + 8 * hi;
            dst[row * DDIM + f] = f2bfu(acc[h][r]);
          }
        } else {  // V, stored transposed [d][key]
          const int col = (nt - 16) * 16 + nlo;
#pragma unroll
          for (int r = 0; r < 8; ++r) {
            int row = mb * 16 + r + 8 * hi;
            VsT[col * LDIM + row] = f2bfu(acc[h][r]);
          }
        }
      }
    }
  }
  __syncthreads();

  // ---- attention: each wave owns 2 query blocks of 16 rows ----
  const float inv_scale = 0.088388347648318447f;  // 1/sqrt(128)
  unsigned short* Ps = Xs + wave * (16 * LDIM);   // per-wave P scratch (Xs recycled)

  for (int qi = 0; qi < 2; ++qi) {
    const int qb = wave * 2 + qi;
    v16bf aQ[4];
#pragma unroll
    for (int kk = 0; kk < 4; ++kk) aQ[kk] = load_a16(Qs, DDIM, qb * 16, kk * 32, lane);

    v8f sacc[16];  // full 16x256 score block in VGPRs
#pragma unroll
    for (int kt = 0; kt < 16; ++kt) {
      v8f acc = v8zero();
#pragma unroll
      for (int kk = 0; kk < 4; ++kk)
        acc = wmma_bf16(aQ[kk], load_bt16(KsT, DDIM, kk * 32, kt * 16, lane), acc);
      const float mv = Mrow[kt * 16 + nlo];
#pragma unroll
      for (int r = 0; r < 8; ++r)
        acc[r] = fminf(fmaxf(acc[r] * inv_scale, -1e4f), 1e4f) - 1e4f * mv;
      sacc[kt] = acc;
    }
    // softmax per row (row = r + 8*hi; columns across 16 tiles x 16 lanes)
#pragma unroll
    for (int r = 0; r < 8; ++r) {
      float mx = -3.4e38f;
#pragma unroll
      for (int kt = 0; kt < 16; ++kt) mx = fmaxf(mx, sacc[kt][r]);
      mx = grp16_max(mx);
      float sum = 0.f;
#pragma unroll
      for (int kt = 0; kt < 16; ++kt) {
        float e = __expf(sacc[kt][r] - mx);
        sacc[kt][r] = e;
        sum += e;
      }
      sum = grp16_sum(sum);
      const float inv = 1.f / sum;
#pragma unroll
      for (int kt = 0; kt < 16; ++kt) sacc[kt][r] *= inv;
    }
    // zero masked keys (post-softmax, as in reference), stash bf16 P
#pragma unroll
    for (int kt = 0; kt < 16; ++kt) {
      const float keep = 1.f - Mrow[kt * 16 + nlo];
#pragma unroll
      for (int r = 0; r < 8; ++r)
        Ps[(r + 8 * hi) * LDIM + kt * 16 + nlo] = f2bfu(sacc[kt][r] * keep);
    }
    // out = P @ V  (P 16x256, V 256x128)
    v16bf aP[8];
#pragma unroll
    for (int kk = 0; kk < 8; ++kk) aP[kk] = load_a16(Ps, LDIM, 0, kk * 32, lane);
#pragma unroll
    for (int nt = 0; nt < 8; ++nt) {
      v8f acc = v8zero();
#pragma unroll
      for (int kk = 0; kk < 8; ++kk)
        acc = wmma_bf16(aP[kk], load_bt16(VsT, LDIM, kk * 32, nt * 16, lane), acc);
#pragma unroll
      for (int r = 0; r < 8; ++r) {
        int row = qb * 16 + r + 8 * hi;
        size_t o = (((size_t)n * LDIM) + row) * (2 * DDIM) + dir * DDIM + nt * 16 + nlo;
        concat_out[o] = f2bfu(acc[r]);
      }
    }
  }
}

// ---------------------------------------------------------------------------
// K3: dense GEMM (concat @ Wd + b), v_keep zeroing, residual, LayerNorm1.
// One block per sequence (256 rows); each wave: 2 M-tiles sharing B loads.
// ---------------------------------------------------------------------------
__global__ __launch_bounds__(256) void dense_ln_kernel(
    const unsigned short* __restrict__ concat, const unsigned short* __restrict__ Wd,
    const float* __restrict__ bias, const float* __restrict__ xp,
    const int* __restrict__ mask, const int* __restrict__ keepflag,
    const float* __restrict__ g1, const float* __restrict__ b1,
    float* __restrict__ vout, unsigned short* __restrict__ vkout) {
  __shared__ unsigned short Ws[DDIM * 2 * DDIM];  // 64KB, [n=128][k=256]
  const int tid = threadIdx.x, lane = tid & 31, wave = tid >> 5;
  const int nlo = lane & 15, hi = (lane >> 4) & 1;
  {  // async stage (pure byte copy -> ASYNCcnt path)
    const unsigned lds0 = (unsigned)(size_t)(void*)Ws;
    const char* g = (const char*)Wd;
    for (int o = tid * 16; o < DDIM * 2 * DDIM * 2; o += 256 * 16)
      async_copy_b128(lds0 + o, g + o);
    wait_async0();
  }
  __syncthreads();

  const int seq = blockIdx.x;  // 512 blocks; one (b,i) sequence each
  const int kv = keepflag[seq];
  const int bb = seq >> 8, ii = seq & 255;

  v16bf aC[2][8];
#pragma unroll
  for (int h = 0; h < 2; ++h) {
    int row0 = seq * 256 + h * 128 + wave * 16;
#pragma unroll
    for (int kk = 0; kk < 8; ++kk) aC[h][kk] = load_a16(concat, 2 * DDIM, row0, kk * 32, lane);
  }
  v8f acc[2][8];
#pragma unroll
  for (int nt = 0; nt < 8; ++nt) {
    float bv = bias[nt * 16 + nlo];
#pragma unroll
    for (int h = 0; h < 2; ++h)
#pragma unroll
      for (int r = 0; r < 8; ++r) acc[h][nt][r] = bv;
#pragma unroll
    for (int kk = 0; kk < 8; ++kk) {
      v16bf bw = load_bt16(Ws, 2 * DDIM, kk * 32, nt * 16, lane);
      acc[0][nt] = wmma_bf16(aC[0][kk], bw, acc[0][nt]);
      acc[1][nt] = wmma_bf16(aC[1][kk], bw, acc[1][nt]);
    }
  }
#pragma unroll
  for (int h = 0; h < 2; ++h) {
    const int row0 = seq * 256 + h * 128 + wave * 16;
#pragma unroll
    for (int nt = 0; nt < 8; ++nt) {
#pragma unroll
      for (int r = 0; r < 8; ++r) {
        size_t pix = (size_t)row0 + r + 8 * hi;
        float v = kv ? acc[h][nt][r] : 0.f;
        acc[h][nt][r] = v + xp[pix * DDIM + nt * 16 + nlo];
      }
    }
#pragma unroll
    for (int r = 0; r < 8; ++r) {
      float s = 0.f, ss = 0.f;
#pragma unroll
      for (int nt = 0; nt < 8; ++nt) { float v = acc[h][nt][r]; s += v; ss += v * v; }
      s = grp16_sum(s);
      ss = grp16_sum(ss);
      float mean = s * (1.f / 128.f);
      float rstd = rsqrtf(ss * (1.f / 128.f) - mean * mean + 1e-5f);
      size_t pix = (size_t)row0 + r + 8 * hi;
      int q = (int)(pix & 255);
      float kp = 1.f - (float)mask[(bb * LDIM + ii) * LDIM + q];
#pragma unroll
      for (int nt = 0; nt < 8; ++nt) {
        int col = nt * 16 + nlo;
        float v = (acc[h][nt][r] - mean) * rstd * g1[col] + b1[col];
        vout[pix * DDIM + col] = v;
        vkout[pix * DDIM + col] = f2bfu(v * kp);
      }
    }
  }
}

// ---------------------------------------------------------------------------
// K4: conv1 (3x3 SAME, implicit GEMM, 288KB tap weights async-staged in LDS)
// + leaky_relu + keep. One block per image row; each wave: 2 pixel tiles.
// ---------------------------------------------------------------------------
__global__ __launch_bounds__(256) void conv1_kernel(const unsigned short* __restrict__ src,
                                                    const unsigned short* __restrict__ Wt,
                                                    const float* __restrict__ bias,
                                                    const int* __restrict__ mask,
                                                    unsigned short* __restrict__ dst) {
  __shared__ unsigned short Ws[9 * DDIM * DDIM];  // 288KB [t][co][ci]
  const int tid = threadIdx.x, lane = tid & 31, wave = tid >> 5;
  const int nlo = lane & 15, hi = (lane >> 4) & 1;
  {
    const unsigned lds0 = (unsigned)(size_t)(void*)Ws;
    const char* g = (const char*)Wt;
    for (int o = tid * 16; o < 9 * DDIM * DDIM * 2; o += 256 * 16)
      async_copy_b128(lds0 + o, g + o);
    wait_async0();
  }
  __syncthreads();

  const int blk = blockIdx.x;  // 512 = B*L
  const int ii = blk & 255, bb = blk >> 8;

  v8f acc[2][8];
#pragma unroll
  for (int nt = 0; nt < 8; ++nt) {
    float bv = bias[nt * 16 + nlo];
#pragma unroll
    for (int h = 0; h < 2; ++h)
#pragma unroll
      for (int r = 0; r < 8; ++r) acc[h][nt][r] = bv;
  }
  for (int t = 0; t < 9; ++t) {
    const int di = t / 3 - 1, dj = t % 3 - 1;
    const int i2 = ii + di;
    if (i2 < 0 || i2 >= LDIM) continue;
    v16bf aT[2][4];
#pragma unroll
    for (int h = 0; h < 2; ++h) {
      const int j2 = h * 128 + wave * 16 + (lane & 15) + dj;
      const bool valid = (j2 >= 0) && (j2 < LDIM);
      const unsigned short* base =
          src + (((size_t)(bb * LDIM + i2)) * LDIM + (valid ? j2 : 0)) * DDIM;
#pragma unroll
      for (int kk = 0; kk < 4; ++kk) aT[h][kk] = load_a16_pred(base, kk * 32, hi, valid);
    }
    const unsigned short* wt = Ws + t * (DDIM * DDIM);
#pragma unroll
    for (int nt = 0; nt < 8; ++nt) {
#pragma unroll
      for (int kk = 0; kk < 4; ++kk) {
        v16bf bw = load_bt16(wt, DDIM, kk * 32, nt * 16, lane);
        acc[0][nt] = wmma_bf16(aT[0][kk], bw, acc[0][nt]);
        acc[1][nt] = wmma_bf16(aT[1][kk], bw, acc[1][nt]);
      }
    }
  }
#pragma unroll
  for (int h = 0; h < 2; ++h) {
#pragma unroll
    for (int r = 0; r < 8; ++r) {
      const int j = h * 128 + wave * 16 + r + 8 * hi;
      const float kp = 1.f - (float)mask[(bb * LDIM + ii) * LDIM + j];
      const size_t pix = ((size_t)(bb * LDIM + ii)) * LDIM + j;
#pragma unroll
      for (int nt = 0; nt < 8; ++nt) {
        float v = acc[h][nt][r];
        v = (v > 0.f) ? v : 0.01f * v;  // leaky_relu
        dst[pix * DDIM + nt * 16 + nlo] = f2bfu(v * kp);
      }
    }
  }
}

// ---------------------------------------------------------------------------
// K5: conv2 + residual + LayerNorm2 -> fp32 output
// ---------------------------------------------------------------------------
__global__ __launch_bounds__(256) void conv2_ln_kernel(const unsigned short* __restrict__ src,
                                                       const unsigned short* __restrict__ Wt,
                                                       const float* __restrict__ bias,
                                                       const float* __restrict__ vres,
                                                       const float* __restrict__ g2,
                                                       const float* __restrict__ b2,
                                                       float* __restrict__ out) {
  __shared__ unsigned short Ws[9 * DDIM * DDIM];
  const int tid = threadIdx.x, lane = tid & 31, wave = tid >> 5;
  const int nlo = lane & 15, hi = (lane >> 4) & 1;
  {
    const unsigned lds0 = (unsigned)(size_t)(void*)Ws;
    const char* g = (const char*)Wt;
    for (int o = tid * 16; o < 9 * DDIM * DDIM * 2; o += 256 * 16)
      async_copy_b128(lds0 + o, g + o);
    wait_async0();
  }
  __syncthreads();

  const int blk = blockIdx.x;
  const int ii = blk & 255, bb = blk >> 8;

  v8f acc[2][8];
#pragma unroll
  for (int nt = 0; nt < 8; ++nt) {
    float bv = bias[nt * 16 + nlo];
#pragma unroll
    for (int h = 0; h < 2; ++h)
#pragma unroll
      for (int r = 0; r < 8; ++r) acc[h][nt][r] = bv;
  }
  for (int t = 0; t < 9; ++t) {
    const int di = t / 3 - 1, dj = t % 3 - 1;
    const int i2 = ii + di;
    if (i2 < 0 || i2 >= LDIM) continue;
    v16bf aT[2][4];
#pragma unroll
    for (int h = 0; h < 2; ++h) {
      const int j2 = h * 128 + wave * 16 + (lane & 15) + dj;
      const bool valid = (j2 >= 0) && (j2 < LDIM);
      const unsigned short* base =
          src + (((size_t)(bb * LDIM + i2)) * LDIM + (valid ? j2 : 0)) * DDIM;
#pragma unroll
      for (int kk = 0; kk < 4; ++kk) aT[h][kk] = load_a16_pred(base, kk * 32, hi, valid);
    }
    const unsigned short* wt = Ws + t * (DDIM * DDIM);
#pragma unroll
    for (int nt = 0; nt < 8; ++nt) {
#pragma unroll
      for (int kk = 0; kk < 4; ++kk) {
        v16bf bw = load_bt16(wt, DDIM, kk * 32, nt * 16, lane);
        acc[0][nt] = wmma_bf16(aT[0][kk], bw, acc[0][nt]);
        acc[1][nt] = wmma_bf16(aT[1][kk], bw, acc[1][nt]);
      }
    }
  }
#pragma unroll
  for (int h = 0; h < 2; ++h) {
#pragma unroll
    for (int r = 0; r < 8; ++r) {
      const int j = h * 128 + wave * 16 + r + 8 * hi;
      const size_t pix = ((size_t)(bb * LDIM + ii)) * LDIM + j;
      float vals[8];
      float s = 0.f, ss = 0.f;
#pragma unroll
      for (int nt = 0; nt < 8; ++nt) {
        float v = acc[h][nt][r] + vres[pix * DDIM + nt * 16 + nlo];
        vals[nt] = v;
        s += v;
        ss += v * v;
      }
      s = grp16_sum(s);
      ss = grp16_sum(ss);
      float mean = s * (1.f / 128.f);
      float rstd = rsqrtf(ss * (1.f / 128.f) - mean * mean + 1e-5f);
#pragma unroll
      for (int nt = 0; nt < 8; ++nt) {
        int col = nt * 16 + nlo;
        out[pix * DDIM + col] = (vals[nt] - mean) * rstd * g2[col] + b2[col];
      }
    }
  }
}

// ---------------------------------------------------------------------------
extern "C" void kernel_launch(void* const* d_in, const int* in_sizes, int n_in,
                              void* d_out, int out_size, void* d_ws, size_t ws_size,
                              hipStream_t stream) {
  (void)in_sizes; (void)n_in; (void)out_size; (void)ws_size;
  const float* x    = (const float*)d_in[0];
  const int*   mask = (const int*)d_in[1];
  const float* pe   = (const float*)d_in[2];
  const float* hqw  = (const float*)d_in[3];
  const float* hqb  = (const float*)d_in[4];
  const float* vqw  = (const float*)d_in[5];
  const float* vqb  = (const float*)d_in[6];
  const float* dw   = (const float*)d_in[7];
  const float* db   = (const float*)d_in[8];
  const float* ln1g = (const float*)d_in[9];
  const float* ln1b = (const float*)d_in[10];
  const float* c1w  = (const float*)d_in[11];
  const float* c1b  = (const float*)d_in[12];
  const float* c2w  = (const float*)d_in[13];
  const float* c2b  = (const float*)d_in[14];
  const float* ln2g = (const float*)d_in[15];
  const float* ln2b = (const float*)d_in[16];
  float* out = (float*)d_out;

  char* ws = (char*)d_ws;
  size_t off = 0;
  float* xp = (float*)(ws + off);                 off += (size_t)16777216 * 4;  // pos-embedded x
  unsigned short* concat = (unsigned short*)(ws + off); off += (size_t)33554432 * 2;  // [hv|vv] bf16
  float* vbuf = (float*)(ws + off);               off += (size_t)16777216 * 4;  // post-LN1 v
  unsigned short* vk = (unsigned short*)(ws + off);   off += (size_t)16777216 * 2;  // v*keep bf16
  unsigned short* c1k = (unsigned short*)(ws + off);  off += (size_t)16777216 * 2;  // conv1 out*keep
  unsigned short* hwb = (unsigned short*)(ws + off);  off += (size_t)49152 * 2;
  unsigned short* vwb = (unsigned short*)(ws + off);  off += (size_t)49152 * 2;
  unsigned short* dwb = (unsigned short*)(ws + off);  off += (size_t)32768 * 2;
  unsigned short* c1bw = (unsigned short*)(ws + off); off += (size_t)147456 * 2;
  unsigned short* c2bw = (unsigned short*)(ws + off); off += (size_t)147456 * 2;
  int* keepflag = (int*)(ws + off);               off += (size_t)512 * 4;

  convert_weights_kernel<<<576, 256, 0, stream>>>(hqw, vqw, dw, c1w, c2w, hwb, vwb, dwb, c1bw, c2bw);
  colsum_kernel<<<2, 256, 0, stream>>>(mask, keepflag);
  prep_kernel<<<65536, 256, 0, stream>>>(x, pe, xp);
  attn_kernel<<<512, 256, 0, stream>>>(xp, mask, hwb, hqb, concat, 0);
  attn_kernel<<<512, 256, 0, stream>>>(xp, mask, vwb, vqb, concat, 1);
  dense_ln_kernel<<<512, 256, 0, stream>>>(concat, dwb, db, xp, mask, keepflag, ln1g, ln1b, vbuf, vk);
  conv1_kernel<<<512, 256, 0, stream>>>(vk, c1bw, c1b, mask, c1k);
  conv2_ln_kernel<<<512, 256, 0, stream>>>(c1k, c2bw, c2b, vbuf, ln2g, ln2b, out);
}
